// SemanticRefinement_21139829031450
// MI455X (gfx1250) — compile-verified, matrix operand order checked by reading, not run on hardware
//
#include <hip/hip_runtime.h>

typedef __attribute__((ext_vector_type(16))) _Float16 v16h;
typedef __attribute__((ext_vector_type(8)))  float    v8f;
typedef __attribute__((ext_vector_type(4)))  unsigned u32x4;
typedef __attribute__((ext_vector_type(4)))  int      i32x4;
typedef __attribute__((ext_vector_type(8)))  int      i32x8;

#define HIDDEN 300
#define NNODES 8192
#define NCLAIM 128

#define HAVE_TDM (__has_builtin(__builtin_amdgcn_tensor_load_to_lds) && \
                  __has_builtin(__builtin_amdgcn_s_wait_tensorcnt))

// ---------------------------------------------------------------------------
// Fragment loaders for v_wmma_f32_16x16x32_f16 (16-bit A/B layout):
// lane = M (A) or N (B); elements 0..7 = K kb..kb+7, 8..15 = K kb+16..kb+23,
// kb = k0 + (lane>>4)*8.
// ---------------------------------------------------------------------------
__device__ __forceinline__ v16h frag_full(const float* __restrict__ row, int kb) {
    const float4 x0 = *(const float4*)(row + kb);
    const float4 x1 = *(const float4*)(row + kb + 4);
    const float4 y0 = *(const float4*)(row + kb + 16);
    const float4 y1 = *(const float4*)(row + kb + 20);
    v16h a;
    a[0]=(_Float16)x0.x; a[1]=(_Float16)x0.y; a[2]=(_Float16)x0.z; a[3]=(_Float16)x0.w;
    a[4]=(_Float16)x1.x; a[5]=(_Float16)x1.y; a[6]=(_Float16)x1.z; a[7]=(_Float16)x1.w;
    a[8]=(_Float16)y0.x; a[9]=(_Float16)y0.y; a[10]=(_Float16)y0.z; a[11]=(_Float16)y0.w;
    a[12]=(_Float16)y1.x; a[13]=(_Float16)y1.y; a[14]=(_Float16)y1.z; a[15]=(_Float16)y1.w;
    return a;
}

__device__ __forceinline__ v16h frag_tail(const float* __restrict__ row, int kb) {
    v16h a;
#pragma unroll
    for (int e = 0; e < 8; ++e) {
        const int k0 = kb + e;
        const int k1 = kb + 16 + e;
        a[e]     = (k0 < HIDDEN) ? (_Float16)row[k0] : (_Float16)0.f;
        a[8 + e] = (k1 < HIDDEN) ? (_Float16)row[k1] : (_Float16)0.f;
    }
    return a;
}

// ---------------------------------------------------------------------------
// Kernel 1: fused score = 0.5*score1 + 0.5*score2
//   score1[n] = H_e[n,:] . W_score1
//   score2[n] = -log( sum_c softmax(H_e H_c^T)[n,c] * rowsum(H_c)[c] + 1e-10 )
// Logits GEMM (8192x128, K=300) via v_wmma_f32_16x16x32_f16.
// Block = 128 threads (4 waves), each wave owns a 16-row M tile.
// ---------------------------------------------------------------------------
__global__ __launch_bounds__(128) void score_kernel(
    const float* __restrict__ He, const float* __restrict__ Hc,
    const float* __restrict__ Ws, float* __restrict__ score)
{
    __shared__ float sSc[NCLAIM];   // rowsum(H_c)
    __shared__ float sW[HIDDEN];    // W_score1
    __shared__ float sS2[64];       // per-row score2

    const int t = threadIdx.x;

    {   // claim row sums (128 threads, one claim each)
        const float* hr = Hc + t * HIDDEN;
        float s = 0.f;
        for (int k = 0; k < HIDDEN; ++k) s += hr[k];
        sSc[t] = s;
    }
    for (int i = t; i < HIDDEN; i += 128) sW[i] = Ws[i];
    __syncthreads();

    const int wave = t >> 5;
    const int lane = t & 31;
    const int kh   = lane >> 4;
    const int l15  = lane & 15;

    const int mbase = blockIdx.x * 64 + wave * 16;
    const float* Arow = He + (long)(mbase + l15) * HIDDEN;

    v8f acc[8];
#pragma unroll
    for (int j = 0; j < 8; ++j) acc[j] = (v8f){0,0,0,0,0,0,0,0};

    // 9 full K steps (K = 0..287), no guards in the hot loop
    for (int ks = 0; ks < 9; ++ks) {
        const int kb = ks * 32 + kh * 8;   // 32B-aligned within the row
        const v16h a = frag_full(Arow, kb);
#pragma unroll
        for (int j = 0; j < 8; ++j) {
            const v16h b = frag_full(Hc + (long)(j * 16 + l15) * HIDDEN, kb);
            acc[j] = __builtin_amdgcn_wmma_f32_16x16x32_f16(
                false, a, false, b, (short)0, acc[j], false, false);
        }
    }
    {   // tail step: K = 288..299, zero-padded to 320
        const int kb = 288 + kh * 8;
        const v16h a = frag_tail(Arow, kb);
#pragma unroll
        for (int j = 0; j < 8; ++j) {
            const v16h b = frag_tail(Hc + (long)(j * 16 + l15) * HIDDEN, kb);
            acc[j] = __builtin_amdgcn_wmma_f32_16x16x32_f16(
                false, a, false, b, (short)0, acc[j], false, false);
        }
    }

    // C layout: M = r + 8*(lane>>4), N = j*16 + (lane&15).
    // Softmax over the 128 claims per row: reduce within the 16-lane half.
#pragma unroll
    for (int r = 0; r < 8; ++r) {
        float mx = -3.4e38f;
#pragma unroll
        for (int j = 0; j < 8; ++j) mx = fmaxf(mx, acc[j][r]);
        for (int off = 1; off < 16; off <<= 1)
            mx = fmaxf(mx, __shfl_xor(mx, off, 32));

        float se = 0.f, swt = 0.f;
#pragma unroll
        for (int j = 0; j < 8; ++j) {
            float e = __expf(acc[j][r] - mx);
            se  += e;
            swt += e * sSc[j * 16 + l15];
        }
        for (int off = 1; off < 16; off <<= 1) {
            se  += __shfl_xor(se,  off, 32);
            swt += __shfl_xor(swt, off, 32);
        }
        if (l15 == 0) {
            float rel = swt / se;           // sum_c p_c * rowsum(H_c)[c]
            sS2[wave * 16 + kh * 8 + r] = -logf(rel + 1e-10f);
        }
    }
    __syncthreads();

    if (t < 64) {
        const int grow = blockIdx.x * 64 + t;
        const float* hr = He + (long)grow * HIDDEN;
        float s1 = 0.f;
        for (int k = 0; k < HIDDEN; ++k) s1 += hr[k] * sW[k];
        score[grow] = 0.5f * s1 + 0.5f * sS2[t];
    }
}

// ---------------------------------------------------------------------------
// Kernel 2: one GGNN step (hidden=1 GRU). h (32 KB) staged into LDS via the
// Tensor Data Mover (1-D D# tile, TENSORcnt tracked); each wave then streams
// one 8192-wide adj row with float4 loads and fuses the GRU pointwise math.
// ---------------------------------------------------------------------------
__global__ __launch_bounds__(256) void ggnn_step(
    const float* __restrict__ adj, const float* __restrict__ hin,
    float* __restrict__ hout,
    const float* __restrict__ wz, const float* __restrict__ uz,
    const float* __restrict__ wr, const float* __restrict__ ur,
    const float* __restrict__ wh, const float* __restrict__ uh)
{
    __shared__ float sh[NNODES];   // 32 KB
    const int t = threadIdx.x;

#if HAVE_TDM
    if (t < 32) {   // wave 0 issues one TDM descriptor for the whole vector
        const unsigned lds_addr = (unsigned)(size_t)sh;   // flat[31:0] == LDS offset
        const unsigned long long ga = (unsigned long long)hin;
        u32x4 g0;
        g0[0] = 1u;                                   // count=1, load, not restore
        g0[1] = lds_addr;                             // lds_addr
        g0[2] = (unsigned)(ga & 0xffffffffu);         // global_addr[31:0]
        g0[3] = (unsigned)(ga >> 32) | (2u << 30);    // global_addr[56:32] | type=2
        i32x8 g1;
        g1[0] = 2 << 16;                 // workgroup_mask=0, data_size=2 (4 bytes)
        g1[1] = (int)(NNODES << 16);     // tensor_dim0[15:0] in bits 63:48
        g1[2] = (int)(1u << 16);         // tensor_dim0[31:16]=0, tensor_dim1=1
        g1[3] = (int)(NNODES << 16);     // tensor_dim1 hi=0, tile_dim0=8192
        g1[4] = 1;                       // tile_dim1=1, tile_dim2=0
        g1[5] = NNODES;                  // tensor_dim0_stride[31:0]
        g1[6] = (int)(NNODES << 16);     // stride0 hi=0, tensor_dim1_stride[15:0]
        g1[7] = 0;                       // tensor_dim1_stride hi
        const i32x4 gz4 = (i32x4){0, 0, 0, 0};
        const i32x8 gz8 = (i32x8){0, 0, 0, 0, 0, 0, 0, 0};
        __builtin_amdgcn_tensor_load_to_lds(g0, g1, gz4, gz4, gz8, 0);
        __builtin_amdgcn_s_wait_tensorcnt(0);
    }
    __syncthreads();
#else
    {
        const float4* h4 = (const float4*)hin;
        float4* s4 = (float4*)sh;
        for (int i = t; i < NNODES / 4; i += 256) s4[i] = h4[i];
        __syncthreads();
    }
#endif

    const int wave = t >> 5;
    const int lane = t & 31;
    const int row  = blockIdx.x * 8 + wave;
    const float* ar = adj + (long)row * NNODES;

    float acc = 0.f;
    for (int c = lane * 4; c < NNODES; c += 128) {
        __builtin_prefetch(ar + c + 1024, 0, 0);
        float4 v = *(const float4*)(ar + c);
        acc += v.x * sh[c] + v.y * sh[c + 1] + v.z * sh[c + 2] + v.w * sh[c + 3];
    }
    for (int off = 16; off; off >>= 1) acc += __shfl_xor(acc, off, 32);

    if (lane == 0) {
        const float a = acc, h = sh[row];
        const float z  = 1.f / (1.f + __expf(-(a * wz[0] + h * uz[0])));
        const float r  = 1.f / (1.f + __expf(-(a * wr[0] + h * ur[0])));
        const float ht = tanhf(a * wh[0] + r * h * uh[0]);
        hout[row] = (1.f - z) * h + z * ht;
    }
}

// ---------------------------------------------------------------------------
// Kernel 3: stable rank selection. rank(n) = #{m: S[m]<S[n]} + #{m<n: S[m]==S[n]}
// keep if rank < n_keep  (== take n_keep smallest with stable tie order)
// ---------------------------------------------------------------------------
__global__ __launch_bounds__(256) void rank_kernel(
    const float* __restrict__ S, int* __restrict__ keep, int limit)
{
    __shared__ float sS[NNODES];   // 32 KB
    const int t = threadIdx.x;
    for (int i = t; i < NNODES; i += 256) sS[i] = S[i];
    __syncthreads();

    const int n = blockIdx.x * 256 + t;
    const float v = sS[n];
    int rank = 0;
    for (int m = 0; m < NNODES; ++m) {
        float u = sS[m];
        rank += (u < v) || (u == v && m < n);
    }
    keep[n] = (rank < limit) ? 1 : 0;
}

// ---------------------------------------------------------------------------
// Kernel 4: exclusive scan of keep flags -> output row positions (1 block).
// ---------------------------------------------------------------------------
__global__ __launch_bounds__(1024) void scan_kernel(
    const int* __restrict__ keep, int* __restrict__ pos)
{
    __shared__ int part[1024];
    const int t = threadIdx.x;
    int f[8];
    int s = 0;
#pragma unroll
    for (int i = 0; i < 8; ++i) { f[i] = keep[t * 8 + i]; s += f[i]; }
    part[t] = s;
    __syncthreads();
    for (int off = 1; off < 1024; off <<= 1) {
        int v = (t >= off) ? part[t - off] : 0;
        __syncthreads();
        part[t] += v;
        __syncthreads();
    }
    int ex = part[t] - s;
#pragma unroll
    for (int i = 0; i < 8; ++i) { pos[t * 8 + i] = ex; ex += f[i]; }
}

// ---------------------------------------------------------------------------
// Kernel 5: gather kept rows + write keep_mask (as 0/1 floats) after them.
// ---------------------------------------------------------------------------
__global__ __launch_bounds__(64) void gather_kernel(
    const float* __restrict__ He, const int* __restrict__ keep,
    const int* __restrict__ pos, float* __restrict__ out, int nkeep)
{
    const int n = blockIdx.x;
    const int t = threadIdx.x;
    if (keep[n]) {
        const float* src = He + (long)n * HIDDEN;
        float* dst = out + (long)pos[n] * HIDDEN;
        for (int k = t; k < HIDDEN; k += 64) dst[k] = src[k];
    }
    if (t == 0) out[(long)nkeep * HIDDEN + n] = keep[n] ? 1.0f : 0.0f;
}

// ---------------------------------------------------------------------------
extern "C" void kernel_launch(void* const* d_in, const int* in_sizes, int n_in,
                              void* d_out, int out_size, void* d_ws, size_t ws_size,
                              hipStream_t stream) {
    const float* He  = (const float*)d_in[0];
    const float* Hc  = (const float*)d_in[1];
    const float* adj = (const float*)d_in[2];
    const float* Ws  = (const float*)d_in[3];
    const float* Wz  = (const float*)d_in[4];
    const float* Uz  = (const float*)d_in[5];
    const float* Wr  = (const float*)d_in[6];
    const float* Ur  = (const float*)d_in[7];
    const float* Wh  = (const float*)d_in[8];
    const float* Uh  = (const float*)d_in[9];
    float* out = (float*)d_out;

    float* h0   = (float*)d_ws;
    float* h1   = h0 + NNODES;
    int*   keep = (int*)(h1 + NNODES);
    int*   pos  = keep + NNODES;

    const int nkeep = (out_size - NNODES) / HIDDEN;   // 5735 for the reference shape

    score_kernel<<<NNODES / 64, 128, 0, stream>>>(He, Hc, Ws, h0);
    ggnn_step<<<NNODES / 8, 256, 0, stream>>>(adj, h0, h1, Wz, Uz, Wr, Ur, Wh, Uh);
    ggnn_step<<<NNODES / 8, 256, 0, stream>>>(adj, h1, h0, Wz, Uz, Wr, Ur, Wh, Uh);
    ggnn_step<<<NNODES / 8, 256, 0, stream>>>(adj, h0, h1, Wz, Uz, Wr, Ur, Wh, Uh);
    rank_kernel<<<NNODES / 256, 256, 0, stream>>>(h1, keep, nkeep);
    scan_kernel<<<1, 1024, 0, stream>>>(keep, pos);
    gather_kernel<<<NNODES, 64, 0, stream>>>(He, keep, pos, out, nkeep);
}
